// MIPNetwork_18614388260884
// MI455X (gfx1250) — matrix operand order repaired and checked by reading, not compiled, not run on hardware
//
#include <hip/hip_runtime.h>
#include <hip/hip_bf16.h>
#include <math.h>

// MIPNetwork for MI455X (gfx1250, wave32, WMMA).
// Dense MLPs on v_wmma_f32_16x16x32_bf16; weights pre-packed once into
// B-fragment layout (bf16) so hot loops load fragments with b128 loads.
// Assumes V and C are multiples of 16 (true for V=C=100000).

#define FM 64

typedef __bf16 bf16_t;
typedef bf16_t v16bf __attribute__((ext_vector_type(16)));
typedef float  v8f   __attribute__((ext_vector_type(8)));

__device__ __forceinline__ v8f wmma_bf16(v16bf a, v16bf b, v8f c) {
  return __builtin_amdgcn_wmma_f32_16x16x32_bf16(false, a, false, b, (short)0, c,
                                                 false, false);
}

__device__ __forceinline__ int lane_id() { return threadIdx.x & 31; }

// ---------------- A fragment loaders (float4-vectorized) --------------------
// A: 16x32 bf16 (MxK). lane L: m = L&15; elems e<8 -> k=kb+off+e,
// e>=8 -> k=kb+off+16+(e-8), off = (L>=16)*8.   (ISA 7.12.2, 16-bit A)
__device__ __forceinline__ void ldA_raw(const float* X, int ld, int rowBase, int kb,
                                        float (&t)[16]) {
  const int lane = lane_id();
  const float* p = X + (size_t)(rowBase + (lane & 15)) * ld + kb + ((lane >> 4) << 3);
  float4 x0 = ((const float4*)p)[0];
  float4 x1 = ((const float4*)p)[1];
  float4 x2 = ((const float4*)(p + 16))[0];
  float4 x3 = ((const float4*)(p + 16))[1];
  t[0]=x0.x; t[1]=x0.y; t[2]=x0.z; t[3]=x0.w; t[4]=x1.x; t[5]=x1.y; t[6]=x1.z; t[7]=x1.w;
  t[8]=x2.x; t[9]=x2.y; t[10]=x2.z; t[11]=x2.w; t[12]=x3.x; t[13]=x3.y; t[14]=x3.z; t[15]=x3.w;
}

__device__ __forceinline__ v16bf ldA_plain(const float* X, int ld, int rowBase, int kb) {
  float t[16]; ldA_raw(X, ld, rowBase, kb, t);
  v16bf a;
#pragma unroll
  for (int e = 0; e < 16; ++e) a[e] = (bf16_t)t[e];
  return a;
}

__device__ __forceinline__ v16bf ldA_relu_scale(const float* X, int ld, int rowBase,
                                                int kb, float s) {
  float t[16]; ldA_raw(X, ld, rowBase, kb, t);
  v16bf a;
#pragma unroll
  for (int e = 0; e < 16; ++e) a[e] = (bf16_t)fmaxf(t[e] * s, 0.0f);
  return a;
}

// value = (X[m][k] - sub[m]) / div[m]; sub may be nullptr
__device__ __forceinline__ v16bf ldA_subdiv(const float* X, int ld, int rowBase, int kb,
                                            const float* sub, const float* div) {
  const int lane = lane_id();
  const int m = rowBase + (lane & 15);
  const float s = sub ? sub[m] : 0.0f;
  const float inv = 1.0f / div[m];
  float t[16]; ldA_raw(X, ld, rowBase, kb, t);
  v16bf a;
#pragma unroll
  for (int e = 0; e < 16; ++e) a[e] = (bf16_t)((t[e] - s) * inv);
  return a;
}

// ---------------- packed B fragments ----------------------------------------
// Fragment fragIdx = kc*nNt + nt, 512 bf16 each: lane L owns 16 contiguous
// bf16 at offset L*16 (one 32B load -> 2x global_load_b128).
__device__ __forceinline__ v16bf ldB_packed(const bf16_t* __restrict__ P, int fragIdx) {
  return *(const v16bf*)(P + ((size_t)fragIdx << 9) + (lane_id() << 4));
}

// Pack row-major f32 W[krows x ldw] (logical K padded to nKc*32, N to nNt*16)
// into B-fragment layout: elem e of lane L in frag (kc,nt) is
// W[kc*32 + (e>=8?16:0) + (L>=16?8:0) + (e&7)][nt*16 + (L&15)], 0 outside bounds.
__global__ void k_packB(const float* __restrict__ Wsrc, int ldw, int krows, int nmax,
                        int nKc, int nNt, bf16_t* __restrict__ dst) {
  int idx = blockIdx.x * blockDim.x + threadIdx.x;
  int total = nKc * nNt * 512;
  if (idx >= total) return;
  int frag = idx >> 9;
  int lane = (idx >> 4) & 31;
  int e = idx & 15;
  int kc = frag / nNt, nt = frag - kc * nNt;
  int k = kc * 32 + ((e >> 3) << 4) + ((lane >> 4) << 3) + (e & 7);
  int n = nt * 16 + (lane & 15);
  float w = (k < krows && n < nmax) ? Wsrc[(size_t)k * ldw + n] : 0.0f;
  dst[idx] = (bf16_t)w;
}

// ---------------- noise (deterministic hash Box-Muller) ---------------------
__device__ __forceinline__ unsigned hash_u32(unsigned x) {
  x ^= x >> 16; x *= 0x7feb352dU; x ^= x >> 15; x *= 0x846ca68bU; x ^= x >> 16;
  return x;
}
__device__ __forceinline__ float noise_val(int pass, int v, int j) {
  unsigned s1 = hash_u32(hash_u32(0x9e3779b9u ^ (unsigned)pass) + (unsigned)v * 4u + (unsigned)j);
  unsigned s2 = hash_u32(s1 ^ 0x85ebca6bu);
  float u1 = ((float)s1 + 1.0f) * (1.0f / 4294967296.0f);
  float u2 = (float)s2 * (1.0f / 4294967296.0f);
  return sqrtf(-2.0f * logf(u1)) * cosf(6.28318530718f * u2);
}

// A-fragment holding noise in k=0..3, zeros elsewhere (var_noisy cols 64..67)
__device__ __forceinline__ v16bf ldA_noise(int pass, int rowBase) {
  const int lane = lane_id();
  v16bf a;
#pragma unroll
  for (int e = 0; e < 16; ++e) a[e] = (bf16_t)0.0f;
  if (lane < 16) {
    const int m = rowBase + lane;
#pragma unroll
    for (int j = 0; j < 4; ++j) a[j] = (bf16_t)noise_val(pass, m, j);
  }
  return a;
}

__device__ __forceinline__ float sigmoidf(float x) { return 1.0f / (1.0f + expf(-x)); }

// ---------------- small utility kernels -------------------------------------
__global__ void k_fill(float* __restrict__ p, long n, float v) {
  long i = (long)blockIdx.x * blockDim.x + threadIdx.x;
  if (i < n) p[i] = v;
}

__global__ void k_add_eps(float* __restrict__ p, int n, float eps) {
  int i = blockIdx.x * blockDim.x + threadIdx.x;
  if (i < n) p[i] += eps;
}

__global__ void k_edge_scalers(const float* __restrict__ ev, const int* __restrict__ econ,
                               const int* __restrict__ evar, float* __restrict__ cs,
                               float* __restrict__ vs, int E) {
  int e = blockIdx.x * blockDim.x + threadIdx.x;
  if (e >= E) return;
  float a = fabsf(ev[e]);
  atomicAdd(&cs[econ[e]], a);
  atomicAdd(&vs[evar[e]], a);
}

__global__ void k_absum(const float* __restrict__ x, float* __restrict__ out, int n) {
  int i = blockIdx.x * blockDim.x + threadIdx.x;
  float v = (i < n) ? fabsf(x[i]) : 0.0f;
#pragma unroll
  for (int o = 16; o > 0; o >>= 1) v += __shfl_xor(v, o, 32);
  if ((threadIdx.x & 31) == 0) atomicAdd(out, v);
}

__global__ void k_obj_norm(const float* __restrict__ obj, const float* __restrict__ sum,
                           float* __restrict__ out, int n) {
  int i = blockIdx.x * blockDim.x + threadIdx.x;
  if (i < n) out[i] = obj[i] / (sum[0] / (float)n + 1e-6f);
}

__global__ void k_cgrad(const float* __restrict__ ev, const int* __restrict__ econ,
                        const int* __restrict__ evar, const float* __restrict__ cs,
                        float* __restrict__ cgr, int E) {
  int e = blockIdx.x * blockDim.x + threadIdx.x;
  if (e >= E) return;
  atomicAdd(&cgr[evar[e]], ev[e] / cs[econ[e]]);
}

// gsum[n] = sum_k vu_w1[128+k][n]  (const_gradient broadcast collapses to rank-1)
__global__ void k_gsum(const float* __restrict__ vu_w1, float* __restrict__ gsum) {
  int n = threadIdx.x;
  if (n < FM) {
    float s = 0.0f;
    for (int k = 0; k < FM; ++k) s += vu_w1[(size_t)(128 + k) * FM + n];
    gsum[n] = s;
  }
}

// ---------------- edge scatter kernels (4 features / thread) ----------------
__global__ void k_scatter_lhs(const float* __restrict__ ev, const int* __restrict__ evar,
                              const int* __restrict__ econ, const float* __restrict__ query,
                              float* __restrict__ lhs, long total16) {
  long idx = (long)blockIdx.x * blockDim.x + threadIdx.x;
  if (idx >= total16) return;
  int e = (int)(idx >> 4), f = (int)(idx & 15) << 2;
  float v = ev[e];
  float4 q = *(const float4*)&query[(size_t)evar[e] * FM + f];
  float* d = &lhs[(size_t)econ[e] * FM + f];
  atomicAdd(d + 0, v * q.x);
  atomicAdd(d + 1, v * q.y);
  atomicAdd(d + 2, v * q.z);
  atomicAdd(d + 3, v * q.w);
}

__global__ void k_scatter_c2v(const float* __restrict__ ev, const int* __restrict__ evar,
                              const int* __restrict__ econ, const float* __restrict__ chi,
                              float* __restrict__ c2v, long total16) {
  long idx = (long)blockIdx.x * blockDim.x + threadIdx.x;
  if (idx >= total16) return;
  int e = (int)(idx >> 4), f = (int)(idx & 15) << 2;
  float v = ev[e];
  float s = (v > 0.0f) ? 1.0f : ((v < 0.0f) ? -1.0f : 0.0f);
  float4 h = *(const float4*)&chi[(size_t)econ[e] * FM + f];
  float* d = &c2v[(size_t)evar[e] * FM + f];
  atomicAdd(d + 0, s * h.x);
  atomicAdd(d + 1, s * h.y);
  atomicAdd(d + 2, s * h.z);
  atomicAdd(d + 3, s * h.w);
}

// ---------------- WMMA GEMM kernels (one 16-row tile per wave32) -------------
// mq stage 1: hidden = relu([vars|noise] @ mq_w1 + b1)   (K=64 + 4 noise)
__global__ void k_mq1(const float* __restrict__ vars, const bf16_t* __restrict__ pw,
                      const float* __restrict__ b1, float* __restrict__ hidden,
                      int ntiles, int pass) {
  const int tile = blockIdx.x * (blockDim.x >> 5) + (threadIdx.x >> 5);
  if (tile >= ntiles) return;
  const int rowBase = tile * 16, lane = lane_id(), hi = lane >> 4, nl = lane & 15;
  v16bf a0 = ldA_plain(vars, FM, rowBase, 0);
  v16bf a1 = ldA_plain(vars, FM, rowBase, 32);
  v16bf an = ldA_noise(pass, rowBase);
#pragma unroll
  for (int t = 0; t < 4; ++t) {
    v8f c = {};
    c = wmma_bf16(a0, ldB_packed(pw, 0 * 4 + t), c);
    c = wmma_bf16(a1, ldB_packed(pw, 1 * 4 + t), c);
    c = wmma_bf16(an, ldB_packed(pw, 2 * 4 + t), c);
    const float bia = b1[t * 16 + nl];
#pragma unroll
    for (int r = 0; r < 8; ++r) {
      int m = rowBase + r + 8 * hi;
      hidden[(size_t)m * FM + t * 16 + nl] = fmaxf(c[r] + bia, 0.0f);
    }
  }
}

// mq stage 2: q = hidden @ mq_w2 + b2; query = blend(sigmoid(q), q, int_mask)
__global__ void k_mq2(const float* __restrict__ hidden, const bf16_t* __restrict__ pw,
                      const float* __restrict__ b2, const int* __restrict__ imask,
                      float* __restrict__ query, int ntiles) {
  const int tile = blockIdx.x * (blockDim.x >> 5) + (threadIdx.x >> 5);
  if (tile >= ntiles) return;
  const int rowBase = tile * 16, lane = lane_id(), hi = lane >> 4, nl = lane & 15;
  v16bf a0 = ldA_plain(hidden, FM, rowBase, 0);
  v16bf a1 = ldA_plain(hidden, FM, rowBase, 32);
#pragma unroll
  for (int t = 0; t < 4; ++t) {
    v8f c = {};
    c = wmma_bf16(a0, ldB_packed(pw, 0 * 4 + t), c);
    c = wmma_bf16(a1, ldB_packed(pw, 1 * 4 + t), c);
    const float bia = b2[t * 16 + nl];
#pragma unroll
    for (int r = 0; r < 8; ++r) {
      int m = rowBase + r + 8 * hi;
      float q = c[r] + bia;
      query[(size_t)m * FM + t * 16 + nl] = imask[m] ? sigmoidf(q) : q;
    }
  }
}

// cu stage 1: h = [constraints | (lhs-b)/cs] @ cu_w1 + b1 (raw); accumulate SS
__global__ void k_cu1(const float* __restrict__ cons, const float* __restrict__ lhs,
                      const float* __restrict__ bvals, const float* __restrict__ cs,
                      const bf16_t* __restrict__ pw, const float* __restrict__ b1,
                      float* __restrict__ hidden, float* __restrict__ ssq, int ntiles) {
  const int tile = blockIdx.x * (blockDim.x >> 5) + (threadIdx.x >> 5);
  if (tile >= ntiles) return;
  const int rowBase = tile * 16, lane = lane_id(), hi = lane >> 4, nl = lane & 15;
  v16bf a0 = ldA_plain(cons, FM, rowBase, 0);
  v16bf a1 = ldA_plain(cons, FM, rowBase, 32);
  v16bf a2 = ldA_subdiv(lhs, FM, rowBase, 0, bvals, cs);
  v16bf a3 = ldA_subdiv(lhs, FM, rowBase, 32, bvals, cs);
  float ss = 0.0f;
#pragma unroll
  for (int t = 0; t < 4; ++t) {
    v8f c = {};
    c = wmma_bf16(a0, ldB_packed(pw, 0 * 4 + t), c);
    c = wmma_bf16(a1, ldB_packed(pw, 1 * 4 + t), c);
    c = wmma_bf16(a2, ldB_packed(pw, 2 * 4 + t), c);
    c = wmma_bf16(a3, ldB_packed(pw, 3 * 4 + t), c);
    const float bia = b1[t * 16 + nl];
#pragma unroll
    for (int r = 0; r < 8; ++r) {
      int m = rowBase + r + 8 * hi;
      float h = c[r] + bia;
      hidden[(size_t)m * FM + t * 16 + nl] = h;
      ss += h * h;
    }
  }
#pragma unroll
  for (int o = 16; o > 0; o >>= 1) ss += __shfl_xor(ss, o, 32);
  if (lane == 0) atomicAdd(ssq, ss);
}

// cu stage 2: tmp = relu(h*scale) @ cu_w2 + b2 (N=128);
// n<64: constraints = tmp + 0.5*constraints; n>=64: const_hi = tmp
__global__ void k_cu2(const float* __restrict__ hidden, const bf16_t* __restrict__ pw,
                      const float* __restrict__ b2, float* __restrict__ cons,
                      float* __restrict__ chi, const float* __restrict__ ssq,
                      int ntiles, int nrows) {
  const int tile = blockIdx.x * (blockDim.x >> 5) + (threadIdx.x >> 5);
  if (tile >= ntiles) return;
  const int rowBase = tile * 16, lane = lane_id(), hi = lane >> 4, nl = lane & 15;
  const float scale = rsqrtf(1e-6f + ssq[0] / (float)nrows);
  v16bf a0 = ldA_relu_scale(hidden, FM, rowBase, 0, scale);
  v16bf a1 = ldA_relu_scale(hidden, FM, rowBase, 32, scale);
#pragma unroll
  for (int t = 0; t < 8; ++t) {
    v8f c = {};
    c = wmma_bf16(a0, ldB_packed(pw, 0 * 8 + t), c);
    c = wmma_bf16(a1, ldB_packed(pw, 1 * 8 + t), c);
    const int n = t * 16 + nl;
    const float bia = b2[n];
#pragma unroll
    for (int r = 0; r < 8; ++r) {
      int m = rowBase + r + 8 * hi;
      float v = c[r] + bia;
      if (n < FM) cons[(size_t)m * FM + n] = v + 0.5f * cons[(size_t)m * FM + n];
      else        chi[(size_t)m * FM + (n - FM)] = v;
    }
  }
}

// vu stage 1: h = [vars | c2v/vs] @ vu_w1[0:128] + cgrad*gsum + obj*w1[192] + b1
__global__ void k_vu1(const float* __restrict__ vars, const float* __restrict__ c2v,
                      const float* __restrict__ vs, const float* __restrict__ cgr,
                      const float* __restrict__ obn, const float* __restrict__ gsum,
                      const bf16_t* __restrict__ pw, const float* __restrict__ w1,
                      const float* __restrict__ b1, float* __restrict__ hidden,
                      float* __restrict__ ssq, int ntiles) {
  const int tile = blockIdx.x * (blockDim.x >> 5) + (threadIdx.x >> 5);
  if (tile >= ntiles) return;
  const int rowBase = tile * 16, lane = lane_id(), hi = lane >> 4, nl = lane & 15;
  v16bf a0 = ldA_plain(vars, FM, rowBase, 0);
  v16bf a1 = ldA_plain(vars, FM, rowBase, 32);
  v16bf a2 = ldA_subdiv(c2v, FM, rowBase, 0, nullptr, vs);
  v16bf a3 = ldA_subdiv(c2v, FM, rowBase, 32, nullptr, vs);
  float ss = 0.0f;
#pragma unroll
  for (int t = 0; t < 4; ++t) {
    v8f c = {};
    c = wmma_bf16(a0, ldB_packed(pw, 0 * 4 + t), c);
    c = wmma_bf16(a1, ldB_packed(pw, 1 * 4 + t), c);
    c = wmma_bf16(a2, ldB_packed(pw, 2 * 4 + t), c);
    c = wmma_bf16(a3, ldB_packed(pw, 3 * 4 + t), c);
    const int n = t * 16 + nl;
    const float gn = gsum[n];
    const float wn = w1[(size_t)192 * FM + n];
    const float bia = b1[n];
#pragma unroll
    for (int r = 0; r < 8; ++r) {
      int m = rowBase + r + 8 * hi;
      float h = c[r] + cgr[m] * gn + obn[m] * wn + bia;
      hidden[(size_t)m * FM + n] = h;
      ss += h * h;
    }
  }
#pragma unroll
  for (int o = 16; o > 0; o >>= 1) ss += __shfl_xor(ss, o, 32);
  if (lane == 0) atomicAdd(ssq, ss);
}

// vu stage 2: variables = relu(h*scale) @ vu_w2 + b2 + 0.5*variables
__global__ void k_vu2(const float* __restrict__ hidden, const bf16_t* __restrict__ pw,
                      const float* __restrict__ b2, float* __restrict__ vars,
                      const float* __restrict__ ssq, int ntiles, int nrows) {
  const int tile = blockIdx.x * (blockDim.x >> 5) + (threadIdx.x >> 5);
  if (tile >= ntiles) return;
  const int rowBase = tile * 16, lane = lane_id(), hi = lane >> 4, nl = lane & 15;
  const float scale = rsqrtf(1e-6f + ssq[0] / (float)nrows);
  v16bf a0 = ldA_relu_scale(hidden, FM, rowBase, 0, scale);
  v16bf a1 = ldA_relu_scale(hidden, FM, rowBase, 32, scale);
#pragma unroll
  for (int t = 0; t < 4; ++t) {
    v8f c = {};
    c = wmma_bf16(a0, ldB_packed(pw, 0 * 4 + t), c);
    c = wmma_bf16(a1, ldB_packed(pw, 1 * 4 + t), c);
    const float bia = b2[t * 16 + nl];
#pragma unroll
    for (int r = 0; r < 8; ++r) {
      int m = rowBase + r + 8 * hi;
      size_t idx = (size_t)m * FM + t * 16 + nl;
      vars[idx] = c[r] + bia + 0.5f * vars[idx];
    }
  }
}

// out stage 1: hidden = relu(vars @ out_w1 + b1)
__global__ void k_out1(const float* __restrict__ vars, const bf16_t* __restrict__ pw,
                       const float* __restrict__ b1, float* __restrict__ hidden,
                       int ntiles) {
  const int tile = blockIdx.x * (blockDim.x >> 5) + (threadIdx.x >> 5);
  if (tile >= ntiles) return;
  const int rowBase = tile * 16, lane = lane_id(), hi = lane >> 4, nl = lane & 15;
  v16bf a0 = ldA_plain(vars, FM, rowBase, 0);
  v16bf a1 = ldA_plain(vars, FM, rowBase, 32);
#pragma unroll
  for (int t = 0; t < 4; ++t) {
    v8f c = {};
    c = wmma_bf16(a0, ldB_packed(pw, 0 * 4 + t), c);
    c = wmma_bf16(a1, ldB_packed(pw, 1 * 4 + t), c);
    const float bia = b1[t * 16 + nl];
#pragma unroll
    for (int r = 0; r < 8; ++r) {
      int m = rowBase + r + 8 * hi;
      hidden[(size_t)m * FM + t * 16 + nl] = fmaxf(c[r] + bia, 0.0f);
    }
  }
}

// out stage 2: q8 = hidden @ out_w2 + b2 (N=8, padded to 16); write d_out
__global__ void k_out2(const float* __restrict__ hidden, const bf16_t* __restrict__ pw,
                       const float* __restrict__ b2, const int* __restrict__ imask,
                       float* __restrict__ out, int ntiles, int V, int pass) {
  const int tile = blockIdx.x * (blockDim.x >> 5) + (threadIdx.x >> 5);
  if (tile >= ntiles) return;
  const int rowBase = tile * 16, lane = lane_id(), hi = lane >> 4, nl = lane & 15;
  v16bf a0 = ldA_plain(hidden, FM, rowBase, 0);
  v16bf a1 = ldA_plain(hidden, FM, rowBase, 32);
  v8f c = {};
  c = wmma_bf16(a0, ldB_packed(pw, 0), c);
  c = wmma_bf16(a1, ldB_packed(pw, 1), c);
  if (nl < 8) {
    const float bia = b2[nl];
#pragma unroll
    for (int r = 0; r < 8; ++r) {
      int m = rowBase + r + 8 * hi;
      float q = c[r] + bia;
      float o = imask[m] ? sigmoidf(q) : q;
      out[((size_t)pass * V + m) * 8 + nl] = o;                        // stacked outputs
      if (pass == 2) out[(size_t)3 * V * 8 + (size_t)m * 8 + nl] = q;  // out_vars
    }
  }
}

// ---------------- host launcher ---------------------------------------------
extern "C" void kernel_launch(void* const* d_in, const int* in_sizes, int n_in,
                              void* d_out, int out_size, void* d_ws, size_t ws_size,
                              hipStream_t stream) {
  const float* edge_val = (const float*)d_in[0];
  const float* bvals    = (const float*)d_in[1];
  const float* obj      = (const float*)d_in[2];
  const float* cu_w1 = (const float*)d_in[3];
  const float* cu_b1 = (const float*)d_in[4];
  const float* cu_w2 = (const float*)d_in[5];
  const float* cu_b2 = (const float*)d_in[6];
  const float* mq_w1 = (const float*)d_in[7];
  const float* mq_b1 = (const float*)d_in[8];
  const float* mq_w2 = (const float*)d_in[9];
  const float* mq_b2 = (const float*)d_in[10];
  const float* vu_w1 = (const float*)d_in[11];
  const float* vu_b1 = (const float*)d_in[12];
  const float* vu_w2 = (const float*)d_in[13];
  const float* vu_b2 = (const float*)d_in[14];
  const float* out_w1 = (const float*)d_in[15];
  const float* out_b1 = (const float*)d_in[16];
  const float* out_w2 = (const float*)d_in[17];
  const float* out_b2 = (const float*)d_in[18];
  const int* evar  = (const int*)d_in[19];
  const int* econ  = (const int*)d_in[20];
  const int* imask = (const int*)d_in[21];
  float* outp = (float*)d_out;

  const int E = in_sizes[0];
  const int C = in_sizes[1];
  const int V = in_sizes[2];
  const int M = (V > C) ? V : C;

  // workspace carve-up (floats)
  float* W = (float*)d_ws;
  size_t o = 0;
  float* vars   = W + o; o += (size_t)V * FM;
  float* cons   = W + o; o += (size_t)C * FM;
  float* hidden = W + o; o += (size_t)M * FM;
  float* query  = W + o; o += (size_t)V * FM;
  float* lhs    = W + o; o += (size_t)C * FM;
  float* chi    = W + o; o += (size_t)C * FM;
  float* c2v    = W + o; o += (size_t)V * FM;
  float* csc    = W + o; o += (size_t)C;
  float* vsc    = W + o; o += (size_t)V;
  float* cgr    = W + o; o += (size_t)V;
  float* obn    = W + o; o += (size_t)V;
  float* gsum   = W + o; o += 64;
  float* scal   = W + o; o += 16;   // [0]=cu SS, [1]=vu SS, [2]=|obj| sum
  o = (o + 15) & ~(size_t)15;       // 64B-align packed-weight area
  bf16_t* pk = (bf16_t*)(W + o);
  size_t po = 0;                    // offsets in bf16 elements (512 per frag)
  bf16_t* p_mq1 = pk + po; po += (size_t)12 * 512;  // K=68->3 chunks, N=64
  bf16_t* p_mq2 = pk + po; po += (size_t)8  * 512;  // 64x64
  bf16_t* p_cu1 = pk + po; po += (size_t)16 * 512;  // 128x64
  bf16_t* p_cu2 = pk + po; po += (size_t)16 * 512;  // 64x128
  bf16_t* p_vu1 = pk + po; po += (size_t)16 * 512;  // vu_w1 rows 0..127
  bf16_t* p_vu2 = pk + po; po += (size_t)8  * 512;  // 64x64
  bf16_t* p_ou1 = pk + po; po += (size_t)8  * 512;  // 64x64
  bf16_t* p_ou2 = pk + po; po += (size_t)2  * 512;  // 64x8 (N padded to 16)

  const int tilesV = V / 16, tilesC = C / 16;
  const int gV = (tilesV + 3) / 4, gC = (tilesC + 3) / 4;  // 4 waves / block
  const long eTotal16 = (long)E * 16;                      // 4 features / thread
  const int gE16 = (int)((eTotal16 + 255) / 256);
  const int gE = (E + 255) / 256;
  auto gb = [](long n) { return (int)((n + 255) / 256); };

  // ----- setup ---------------------------------------------------------------
  k_fill<<<gb((long)V * FM), 256, 0, stream>>>(vars, (long)V * FM, 1.0f);
  k_fill<<<gb((long)C * FM), 256, 0, stream>>>(cons, (long)C * FM, 1.0f);
  k_fill<<<gb(C), 256, 0, stream>>>(csc, C, 0.0f);
  k_fill<<<gb(V), 256, 0, stream>>>(vsc, V, 0.0f);
  k_fill<<<gb(V), 256, 0, stream>>>(cgr, V, 0.0f);
  k_fill<<<1, 32, 0, stream>>>(scal, 16, 0.0f);
  k_edge_scalers<<<gE, 256, 0, stream>>>(edge_val, econ, evar, csc, vsc, E);
  k_add_eps<<<gb(C), 256, 0, stream>>>(csc, C, 1e-6f);
  k_add_eps<<<gb(V), 256, 0, stream>>>(vsc, V, 1e-6f);
  k_absum<<<gb(V), 256, 0, stream>>>(obj, scal + 2, V);
  k_obj_norm<<<gb(V), 256, 0, stream>>>(obj, scal + 2, obn, V);
  k_cgrad<<<gE, 256, 0, stream>>>(edge_val, econ, evar, csc, cgr, E);
  k_gsum<<<1, 64, 0, stream>>>(vu_w1, gsum);

  // pack all weights into B-fragment bf16 layout (once)
  k_packB<<<gb(12 * 512), 256, 0, stream>>>(mq_w1, 64, 68, 64, 3, 4, p_mq1);
  k_packB<<<gb(8 * 512),  256, 0, stream>>>(mq_w2, 64, 64, 64, 2, 4, p_mq2);
  k_packB<<<gb(16 * 512), 256, 0, stream>>>(cu_w1, 64, 128, 64, 4, 4, p_cu1);
  k_packB<<<gb(16 * 512), 256, 0, stream>>>(cu_w2, 128, 64, 128, 2, 8, p_cu2);
  k_packB<<<gb(16 * 512), 256, 0, stream>>>(vu_w1, 64, 128, 64, 4, 4, p_vu1);
  k_packB<<<gb(8 * 512),  256, 0, stream>>>(vu_w2, 64, 64, 64, 2, 4, p_vu2);
  k_packB<<<gb(8 * 512),  256, 0, stream>>>(out_w1, 64, 64, 64, 2, 4, p_ou1);
  k_packB<<<gb(2 * 512),  256, 0, stream>>>(out_w2, 8, 64, 8, 2, 1, p_ou2);

  // ----- passes --------------------------------------------------------------
  for (int i = 0; i < 3; ++i) {
    k_fill<<<gb((long)C * FM), 256, 0, stream>>>(lhs, (long)C * FM, 0.0f);
    k_fill<<<gb((long)V * FM), 256, 0, stream>>>(c2v, (long)V * FM, 0.0f);
    k_fill<<<1, 32, 0, stream>>>(scal, 2, 0.0f);

    k_mq1<<<gV, 128, 0, stream>>>(vars, p_mq1, mq_b1, hidden, tilesV, i);
    k_mq2<<<gV, 128, 0, stream>>>(hidden, p_mq2, mq_b2, imask, query, tilesV);
    k_scatter_lhs<<<gE16, 256, 0, stream>>>(edge_val, evar, econ, query, lhs, eTotal16);
    k_cu1<<<gC, 128, 0, stream>>>(cons, lhs, bvals, csc, p_cu1, cu_b1, hidden,
                                  scal + 0, tilesC);
    k_cu2<<<gC, 128, 0, stream>>>(hidden, p_cu2, cu_b2, cons, chi, scal + 0,
                                  tilesC, C);
    k_scatter_c2v<<<gE16, 256, 0, stream>>>(edge_val, evar, econ, chi, c2v, eTotal16);
    k_vu1<<<gV, 128, 0, stream>>>(vars, c2v, vsc, cgr, obn, gsum, p_vu1, vu_w1,
                                  vu_b1, hidden, scal + 1, tilesV);
    k_vu2<<<gV, 128, 0, stream>>>(hidden, p_vu2, vu_b2, vars, scal + 1, tilesV, V);
    k_out1<<<gV, 128, 0, stream>>>(vars, p_ou1, out_b1, hidden, tilesV);
    k_out2<<<gV, 128, 0, stream>>>(hidden, p_ou2, out_b2, imask, outp, tilesV, V, i);
  }
}
